// CustomGRU_19061064860065
// MI455X (gfx1250) — matrix operand order repaired and checked by reading, not compiled, not run on hardware
//
#include <hip/hip_runtime.h>
#include <stdint.h>

typedef __attribute__((ext_vector_type(16))) __bf16 v16bf;
typedef __attribute__((ext_vector_type(8)))  __bf16 v8bf;
typedef __attribute__((ext_vector_type(8)))  float  v8f;

// ---------- bf16 helpers (round-to-nearest-even) ----------
__device__ __forceinline__ uint16_t f2b(float f) {
    unsigned u = __float_as_uint(f);
    unsigned r = u + 0x7fffu + ((u >> 16) & 1u);
    return (uint16_t)(r >> 16);
}
__device__ __forceinline__ float b2f(uint16_t h) {
    return __uint_as_float(((unsigned)h) << 16);
}
__device__ __forceinline__ float sigmoidf_(float x) {
    return 1.0f / (1.0f + __expf(-x));
}

// ---------- WMMA fragment loaders (bf16, 16x16x32) ----------
// A-matrix 16x32 (MxK), row-major source with leading dim ld (elements).
// lane<16: row=lane, K=k0..k0+15 contiguous; lane>=16: row=lane-16, K=k0+16..k0+31
__device__ __forceinline__ v16bf load_a_frag(const uint16_t* Abase, int ld, int k0, int lane) {
    int r    = lane & 15;
    int koff = k0 + ((lane & 16) ? 16 : 0);
    const uint16_t* p = Abase + (size_t)r * ld + koff;
    v8bf lo = *(const v8bf*)(p);
    v8bf hi = *(const v8bf*)(p + 8);
    v16bf out;
#pragma unroll
    for (int i = 0; i < 8; ++i) { out[i] = lo[i]; out[8 + i] = hi[i]; }
    return out;
}
// B-matrix 32x16 (KxN), sourced from pre-transposed BT (N x K) row-major, leading dim ld=K.
// lane<16: col n=lane, K=k0..k0+15 contiguous; lane>=16: col n=lane-16, K=k0+16..k0+31
__device__ __forceinline__ v16bf load_b_frag(const uint16_t* BTbase, int ld, int k0, int lane) {
    int n    = lane & 15;
    int koff = k0 + ((lane & 16) ? 16 : 0);
    return *(const v16bf*)(BTbase + (size_t)n * ld + koff);
}

#define WMMA_BF16(a, b, c) \
    __builtin_amdgcn_wmma_f32_16x16x32_bf16(false, (a), false, (b), (short)0, (c), false, false)

// ---------- conversion / init kernels ----------
__global__ void k_f32_to_bf16(const float* __restrict__ x, uint16_t* __restrict__ y, int n) {
    int i = blockIdx.x * blockDim.x + threadIdx.x;
    if (i < n) y[i] = f2b(x[i]);
}
// W is (K x N) f32 row-major -> WT (N x K) bf16 row-major
__global__ void k_transpose_bf16(const float* __restrict__ W, uint16_t* __restrict__ WT,
                                 int K, int N) {
    int i = blockIdx.x * blockDim.x + threadIdx.x;
    if (i < K * N) {
        int k = i / N, n = i - k * N;
        WT[(size_t)n * K + k] = f2b(W[i]);
    }
}
__global__ void k_zero_bf16(uint16_t* __restrict__ p, int n) {
    int i = blockIdx.x * blockDim.x + threadIdx.x;
    if (i < n) p[i] = 0;
}

// ---------- big input-projection GEMM: C(MxN) = A(MxK)@B(KxN) + bias ----------
// A: bf16 row-major, BT: bf16 (N x K), C: f32. Each wave computes a 16x64 tile
// (4 accumulators, A-fragment reused 4x). M%16==0, N%64==0, K%32==0.
__global__ void k_wmma_gemm_bias(const uint16_t* __restrict__ A, const uint16_t* __restrict__ BT,
                                 const float* __restrict__ bias, float* __restrict__ C,
                                 int M, int N, int K) {
    int lane = threadIdx.x & 31;
    int wid  = (blockIdx.x * blockDim.x + threadIdx.x) >> 5;
    int ng4  = N >> 6;
    int mt   = wid / ng4;
    int ng   = wid - mt * ng4;
    if (mt >= (M >> 4)) return;
    const uint16_t* Ab  = A  + (size_t)mt * 16 * K;
    const uint16_t* Bb0 = BT + (size_t)(ng * 64 +  0) * K;
    const uint16_t* Bb1 = BT + (size_t)(ng * 64 + 16) * K;
    const uint16_t* Bb2 = BT + (size_t)(ng * 64 + 32) * K;
    const uint16_t* Bb3 = BT + (size_t)(ng * 64 + 48) * K;
    v8f c0 = {}, c1 = {}, c2 = {}, c3 = {};
    for (int k0 = 0; k0 < K; k0 += 32) {
        v16bf a  = load_a_frag(Ab,  K, k0, lane);
        v16bf b0 = load_b_frag(Bb0, K, k0, lane);
        v16bf b1 = load_b_frag(Bb1, K, k0, lane);
        v16bf b2 = load_b_frag(Bb2, K, k0, lane);
        v16bf b3 = load_b_frag(Bb3, K, k0, lane);
        c0 = WMMA_BF16(a, b0, c0);
        c1 = WMMA_BF16(a, b1, c1);
        c2 = WMMA_BF16(a, b2, c2);
        c3 = WMMA_BF16(a, b3, c3);
    }
    int colL  = lane & 15;
    int col0  = ng * 64 + colL;
    int mbase = mt * 16 + ((lane & 16) ? 8 : 0);
    float bz0 = bias[col0], bz1 = bias[col0 + 16], bz2 = bias[col0 + 32], bz3 = bias[col0 + 48];
#pragma unroll
    for (int i = 0; i < 8; ++i) {
        float* row = C + (size_t)(mbase + i) * N;
        row[col0]      = c0[i] + bz0;
        row[col0 + 16] = c1[i] + bz1;
        row[col0 + 32] = c2[i] + bz2;
        row[col0 + 48] = c3[i] + bz3;
    }
}

// ---------- fused GRU recurrent step ----------
// rec = h(128x512) @ Urec(512x1536) via UT(1536x512) bf16; then gates + state update.
// gx: f32 (B*T x 1536), rows indexed b*T + t. hin/hout: bf16 (128x512) ping-pong.
// hseq (nullable): bf16 (B x T x 512). 256 waves: mt=0..7, nt=0..31.
__global__ void k_gru_step(const uint16_t* __restrict__ hin, const uint16_t* __restrict__ UT,
                           const float* __restrict__ gx, const float* __restrict__ br,
                           uint16_t* __restrict__ hout, uint16_t* __restrict__ hseq,
                           int t, int T) {
    const int Udim = 512, G = 1536;
    int lane = threadIdx.x & 31;
    int wid  = (blockIdx.x * blockDim.x + threadIdx.x) >> 5;
    int nt   = wid & 31;
    int mt   = wid >> 5;
    const uint16_t* Ab = hin + (size_t)mt * 16 * Udim;
    const uint16_t* Bz = UT + (size_t)(nt * 16)        * Udim;
    const uint16_t* Br = UT + (size_t)(512  + nt * 16) * Udim;
    const uint16_t* Bh = UT + (size_t)(1024 + nt * 16) * Udim;
    v8f az = {}, ar = {}, ah = {};
#pragma unroll 4
    for (int k0 = 0; k0 < 512; k0 += 32) {
        v16bf a  = load_a_frag(Ab, Udim, k0, lane);
        v16bf bz = load_b_frag(Bz, Udim, k0, lane);
        v16bf brf = load_b_frag(Br, Udim, k0, lane);
        v16bf bh = load_b_frag(Bh, Udim, k0, lane);
        az = WMMA_BF16(a, bz,  az);
        ar = WMMA_BF16(a, brf, ar);
        ah = WMMA_BF16(a, bh,  ah);
    }
    int u = nt * 16 + (lane & 15);
    float brz = br[u], brr = br[512 + u], brh = br[1024 + u];
    int mbase = mt * 16 + ((lane & 16) ? 8 : 0);
#pragma unroll
    for (int i = 0; i < 8; ++i) {
        int b = mbase + i;
        const float* g = gx + ((size_t)b * T + t) * G;
        float z  = sigmoidf_(g[u]        + az[i] + brz);
        float r  = sigmoidf_(g[512 + u]  + ar[i] + brr);
        float hh = fmaxf(0.0f, g[1024 + u] + r * (ah[i] + brh));
        float hold = b2f(hin[(size_t)b * Udim + u]);
        float hnew = z * hold + (1.0f - z) * hh;
        uint16_t hb = f2b(hnew);
        hout[(size_t)b * Udim + u] = hb;
        if (hseq) hseq[((size_t)b * T + t) * Udim + u] = hb;
    }
}

// ---------- dense head: out[b] = sum_u h2[b,u]*Wd[u] + bd ----------
__global__ void k_head(const uint16_t* __restrict__ h2, const float* __restrict__ Wd,
                       const float* __restrict__ bd, float* __restrict__ out) {
    int b = blockIdx.x * blockDim.x + threadIdx.x;
    if (b >= 128) return;
    float s = 0.0f;
    for (int u = 0; u < 512; ++u) s += b2f(h2[(size_t)b * 512 + u]) * Wd[u];
    out[b] = s + bd[0];
}

extern "C" void kernel_launch(void* const* d_in, const int* in_sizes, int n_in,
                              void* d_out, int out_size, void* d_ws, size_t ws_size,
                              hipStream_t stream) {
    (void)in_sizes; (void)n_in; (void)out_size; (void)ws_size;
    const int B = 128, T = 512, F = 128, U = 512, G = 3 * U;

    const float* x   = (const float*)d_in[0];
    const float* W1  = (const float*)d_in[1];
    const float* U1  = (const float*)d_in[2];
    const float* bi1 = (const float*)d_in[3];
    const float* br1 = (const float*)d_in[4];
    const float* W2  = (const float*)d_in[5];
    const float* U2  = (const float*)d_in[6];
    const float* bi2 = (const float*)d_in[7];
    const float* br2 = (const float*)d_in[8];
    const float* Wd  = (const float*)d_in[9];
    const float* bd  = (const float*)d_in[10];
    float* out = (float*)d_out;

    // workspace carve-up (256B aligned)
    char* ws = (char*)d_ws;
    size_t off = 0;
    auto carve = [&](size_t bytes) -> void* {
        void* p = ws + off;
        off = (off + bytes + 255) & ~(size_t)255;
        return p;
    };
    float*    gx    = (float*)   carve((size_t)B * T * G * sizeof(float));    // 402 MB, reused
    uint16_t* xb    = (uint16_t*)carve((size_t)B * T * F * 2);                // x in bf16
    uint16_t* h1seq = (uint16_t*)carve((size_t)B * T * U * 2);                // layer-1 outputs
    uint16_t* W1T   = (uint16_t*)carve((size_t)G * F * 2);
    uint16_t* U1T   = (uint16_t*)carve((size_t)G * U * 2);
    uint16_t* W2T   = (uint16_t*)carve((size_t)G * U * 2);
    uint16_t* U2T   = (uint16_t*)carve((size_t)G * U * 2);
    uint16_t* h1A   = (uint16_t*)carve((size_t)B * U * 2);
    uint16_t* h1B   = (uint16_t*)carve((size_t)B * U * 2);
    uint16_t* h2A   = (uint16_t*)carve((size_t)B * U * 2);
    uint16_t* h2B   = (uint16_t*)carve((size_t)B * U * 2);

    // --- one-time-per-call conversions / init ---
    {
        int n = B * T * F;
        k_f32_to_bf16<<<(n + 255) / 256, 256, 0, stream>>>(x, xb, n);
    }
    k_transpose_bf16<<<(F * G + 255) / 256, 256, 0, stream>>>(W1, W1T, F, G);
    k_transpose_bf16<<<(U * G + 255) / 256, 256, 0, stream>>>(U1, U1T, U, G);
    k_transpose_bf16<<<(U * G + 255) / 256, 256, 0, stream>>>(W2, W2T, U, G);
    k_transpose_bf16<<<(U * G + 255) / 256, 256, 0, stream>>>(U2, U2T, U, G);
    k_zero_bf16<<<(B * U + 255) / 256, 256, 0, stream>>>(h1A, B * U);
    k_zero_bf16<<<(B * U + 255) / 256, 256, 0, stream>>>(h2A, B * U);

    // --- layer 1: input projection (M=65536, K=128, N=1536) ---
    {
        int M = B * T;
        int waves = (M / 16) * (G / 64);           // 98304
        k_wmma_gemm_bias<<<waves / 8, 256, 0, stream>>>(xb, W1T, bi1, gx, M, G, F);
    }
    // --- layer 1: recurrence ---
    for (int t = 0; t < T; ++t) {
        const uint16_t* hin = (t & 1) ? h1B : h1A;
        uint16_t*       ho  = (t & 1) ? h1A : h1B;
        k_gru_step<<<64, 128, 0, stream>>>(hin, U1T, gx, br1, ho, h1seq, t, T);
    }
    // --- layer 2: input projection (M=65536, K=512, N=1536), gx reused ---
    {
        int M = B * T;
        int waves = (M / 16) * (G / 64);
        k_wmma_gemm_bias<<<waves / 8, 256, 0, stream>>>(h1seq, W2T, bi2, gx, M, G, U);
    }
    // --- layer 2: recurrence (no sequence output) ---
    for (int t = 0; t < T; ++t) {
        const uint16_t* hin = (t & 1) ? h2B : h2A;
        uint16_t*       ho  = (t & 1) ? h2A : h2B;
        k_gru_step<<<64, 128, 0, stream>>>(hin, U2T, gx, br2, ho, (uint16_t*)nullptr, t, T);
    }
    // T=512 even -> final state in h2A
    k_head<<<1, 128, 0, stream>>>(h2A, Wd, bd, out);
}
